// PolicyNetwork_43817256354107
// MI455X (gfx1250) — matrix operand order repaired and checked by reading, not compiled, not run on hardware
//
#include <hip/hip_runtime.h>
#include <cstdint>
#include <cstddef>

#define Bsz 4096
#define Ssz 16
#define Gsz 128
#define Esz 256
#define Hsz 512
#define Psz 16

typedef __bf16 bf16x16 __attribute__((ext_vector_type(16)));
typedef float  f32x8   __attribute__((ext_vector_type(8)));

union alignas(32) Frag {
    bf16x16 v;
    uint4   q[2];
};

__device__ inline unsigned short f2bf(float f) {
    unsigned u = __builtin_bit_cast(unsigned, f);
    unsigned r = u + 0x7fffu + ((u >> 16) & 1u);   // round-to-nearest-even
    return (unsigned short)(r >> 16);
}

__device__ inline float sigm(float x) { return 1.0f / (1.0f + __expf(-x)); }

// ---------------------------------------------------------------------------
// Generic bf16 GEMM with fp32 accumulate via v_wmma_f32_16x16x32_bf16.
//   C[M,N] = A[M,K](lda) @ W[N,K]^T + bias
// Block: 256 threads (8 wave32), tile 64(M) x 256(N); wave tile 32x64.
// Requires M%64==0, N%256==0, K%32==0 (true for all shapes here).
// ---------------------------------------------------------------------------
template <bool OUT_BF16>
__global__ void gemm_bf16(float* __restrict__ Cf, unsigned short* __restrict__ Cb,
                          const unsigned short* __restrict__ A, int lda,
                          const unsigned short* __restrict__ W,
                          const float* __restrict__ bias,
                          int N, int K) {
    const int tid   = threadIdx.x;
    const int wid   = tid >> 5;
    const int lane  = tid & 31;
    const int lm    = lane & 15;       // row/col within 16
    const int lh    = lane >> 4;       // half selector
    const int waveM = wid & 1;
    const int waveN = wid >> 1;
    const int mBase = blockIdx.x * 64 + waveM * 32;
    const int nBase = blockIdx.y * 256 + waveN * 64;

    f32x8 acc[2][4];
#pragma unroll
    for (int i = 0; i < 2; ++i)
#pragma unroll
        for (int j = 0; j < 4; ++j)
#pragma unroll
            for (int v = 0; v < 8; ++v) acc[i][j][v] = 0.0f;

    const unsigned short* aRow[2];
#pragma unroll
    for (int i = 0; i < 2; ++i)
        aRow[i] = A + (size_t)(mBase + i * 16 + lm) * (size_t)lda;
    const unsigned short* wRow[4];
#pragma unroll
    for (int j = 0; j < 4; ++j)
        wRow[j] = W + (size_t)(nBase + j * 16 + lm) * (size_t)K;

    for (int k0 = 0; k0 < K; k0 += 32) {
        Frag fa[2];
#pragma unroll
        for (int i = 0; i < 2; ++i) {
            // A 16x32 bf16 layout: lanes 0-15 -> K {0..7,16..23}; lanes 16-31 -> K {8..15,24..31}
            fa[i].q[0] = *(const uint4*)(aRow[i] + k0 + lh * 8);
            fa[i].q[1] = *(const uint4*)(aRow[i] + k0 + 16 + lh * 8);
        }
        Frag fb[4];
#pragma unroll
        for (int j = 0; j < 4; ++j) {
            // B 32x16 bf16: lane = column (weight row), lanes 0-15 K {0..7,16..23}, lanes 16-31 K {8..15,24..31}
            const uint4* wq = (const uint4*)(wRow[j] + k0 + lh * 8);
            fb[j].q[0] = wq[0];
            fb[j].q[1] = wq[2];
        }
#pragma unroll
        for (int i = 0; i < 2; ++i)
#pragma unroll
            for (int j = 0; j < 4; ++j)
                acc[i][j] = __builtin_amdgcn_wmma_f32_16x16x32_bf16(
                    false, fa[i].v, false, fb[j].v, (short)0, acc[i][j], false, false);
    }

#pragma unroll
    for (int i = 0; i < 2; ++i)
#pragma unroll
        for (int j = 0; j < 4; ++j) {
            const int col  = nBase + j * 16 + lm;
            const int row0 = mBase + i * 16 + lh * 8;
            const float bv = bias ? bias[col] : 0.0f;
#pragma unroll
            for (int v = 0; v < 8; ++v) {
                const float val = acc[i][j][v] + bv;
                const size_t idx = (size_t)(row0 + v) * (size_t)N + col;
                if constexpr (OUT_BF16) Cb[idx] = f2bf(val);
                else                    Cf[idx] = val;
            }
        }
}

// ---------------------------------------------------------------------------
// fp32 -> bf16 convert with optional repacking (dst row stride / offset).
// ---------------------------------------------------------------------------
__global__ void cvt_pack(const float* __restrict__ src, unsigned short* __restrict__ dst,
                         int rows, int cols, int dstStride, int dstOff) {
    int idx = blockIdx.x * blockDim.x + threadIdx.x;
    if (idx >= rows * cols) return;
    int r = idx / cols, c = idx - r * cols;
    dst[(size_t)r * dstStride + dstOff + c] = f2bf(src[idx]);
}

__global__ void zero_f32(float* p, int n) {
    int i = blockIdx.x * blockDim.x + threadIdx.x;
    if (i < n) p[i] = 0.0f;
}
__global__ void zero_u16(unsigned short* p, int n) {
    int i = blockIdx.x * blockDim.x + threadIdx.x;
    if (i < n) p[i] = 0;
}
__global__ void init_out(float* p) {
    int i = blockIdx.x * blockDim.x + threadIdx.x;
    if (i < Bsz * Psz + Bsz) p[i] = (i < Bsz * Psz) ? -1.0f : 0.0f;
}

// Copy emb[:, t, :] (bf16) into xh[:, 0:256]; 16B per thread.
__global__ void copy_emb_xh(const unsigned short* __restrict__ emb,
                            unsigned short* __restrict__ xh, int t) {
    int idx = blockIdx.x * blockDim.x + threadIdx.x;
    if (idx >= Bsz * 32) return;
    int b = idx >> 5, q = idx & 31;
    ((uint4*)(xh + (size_t)b * 768))[q] =
        ((const uint4*)(emb + ((size_t)b * Ssz + t) * Esz))[q];
}

// Broadcast dec_input0 into xh[:, 0:256].
__global__ void fill_dec0(const float* __restrict__ d0, unsigned short* __restrict__ xh) {
    int idx = blockIdx.x * blockDim.x + threadIdx.x;
    if (idx >= Bsz * Esz) return;
    int b = idx >> 8, e = idx & 255;
    xh[(size_t)b * 768 + e] = f2bf(d0[e]);
}

// LSTM pointwise: consumes gates (B x 2048, order i,f,g,o), updates c, writes
// bf16(h) into xh[:,256:768] and (encoder only) into ref[:, t, :].
__global__ void lstm_pw(const float* __restrict__ gates, float* __restrict__ c,
                        unsigned short* __restrict__ xh, unsigned short* __restrict__ ref,
                        int t) {
    int idx = blockIdx.x * blockDim.x + threadIdx.x;   // < B*H
    if (idx >= Bsz * Hsz) return;
    int b = idx >> 9, j = idx & 511;
    const float* g = gates + (size_t)b * 2048;
    float gi = g[j], gf = g[512 + j], gg = g[1024 + j], go = g[1536 + j];
    float cn = sigm(gf) * c[idx] + sigm(gi) * tanhf(gg);
    float h  = sigm(go) * tanhf(cn);
    c[idx] = cn;
    unsigned short hb = f2bf(h);
    xh[(size_t)b * 768 + 256 + j] = hb;
    if (t >= 0) ref[((size_t)b * Ssz + t) * Hsz + j] = hb;
}

// Pointer/attention decode step: one wave32 per batch row.
// logits[s] = Vec2 . (CLIP*tanh(qp + u2[s])) - MASK_PENALTY*mask[s]
// -> log_softmax, argmax, ll accumulation, mask update, din gather, mapping.
__global__ void attn_step(const float* __restrict__ qp, const float* __restrict__ u2,
                          const float* __restrict__ Vec2, float* __restrict__ mask,
                          const unsigned short* __restrict__ emb,
                          unsigned short* __restrict__ xh,
                          float* __restrict__ outMap, float* __restrict__ outLL, int p) {
    const int wid = threadIdx.x >> 5;
    const int lane = threadIdx.x & 31;
    const int b = blockIdx.x * 8 + wid;

    float qv[16], vv[16];
    const float* qrow = qp + (size_t)b * Hsz;
#pragma unroll
    for (int j = 0; j < 16; ++j) {
        qv[j] = qrow[lane * 16 + j];
        vv[j] = Vec2[lane * 16 + j];
    }

    float mylogit = -3.0e38f;
    for (int s = 0; s < Ssz; ++s) {
        const float* urow = u2 + ((size_t)b * Ssz + s) * Hsz;
        float acc = 0.0f;
#pragma unroll
        for (int j = 0; j < 16; ++j)
            acc += vv[j] * (10.0f * tanhf(qv[j] + urow[lane * 16 + j]));
        for (int off = 16; off; off >>= 1) acc += __shfl_xor(acc, off, 32);
        acc -= 1.0e8f * mask[b * Ssz + s];
        if (lane == s) mylogit = acc;   // lanes 0..15 collect the 16 logits
    }

    // softmax statistics over the 16 logit-holding lanes
    float m = mylogit;
    for (int off = 8; off; off >>= 1) m = fmaxf(m, __shfl_xor(m, off, 16));
    float e = __expf(mylogit - m);
    float se = e;
    for (int off = 8; off; off >>= 1) se += __shfl_xor(se, off, 16);
    float logZ = m + __logf(se);

    // argmax with lowest-index tie break
    float av = mylogit;
    int   ai = lane & 15;
    for (int off = 8; off; off >>= 1) {
        float ov = __shfl_xor(av, off, 16);
        int   oi = __shfl_xor(ai, off, 16);
        if (ov > av || (ov == av && oi < ai)) { av = ov; ai = oi; }
    }
    const int nxt = __shfl(ai, 0, 32);
    const float lp = __shfl(mylogit, nxt, 32) - __shfl(logZ, 0, 32);

    if (lane == 0) {
        outLL[b] += lp;
        mask[b * Ssz + nxt] += 1.0f;
        outMap[b * Psz + nxt] = (float)(p >> 2);   // nodes = repeat(arange(4),4)
    }
    // din = emb[b, nxt, :] -> xh[:,0:256]  (512 B = 32 lanes x 16 B)
    const uint4* src = (const uint4*)(emb + ((size_t)b * Ssz + nxt) * Esz);
    uint4* dst = (uint4*)(xh + (size_t)b * 768);
    dst[lane] = src[lane];
}

// ---------------------------------------------------------------------------

static inline size_t alignup(size_t x) { return (x + 255) & ~(size_t)255; }

extern "C" void kernel_launch(void* const* d_in, const int* in_sizes, int n_in,
                              void* d_out, int out_size, void* d_ws, size_t ws_size,
                              hipStream_t stream) {
    const float* x        = (const float*)d_in[0];
    const float* emb_W    = (const float*)d_in[1];
    const float* enc_Wih  = (const float*)d_in[2];
    const float* enc_Whh  = (const float*)d_in[3];
    const float* enc_b    = (const float*)d_in[4];
    const float* dec_Wih  = (const float*)d_in[5];
    const float* dec_Whh  = (const float*)d_in[6];
    const float* dec_b    = (const float*)d_in[7];
    const float* Wq2      = (const float*)d_in[8];
    const float* bq2      = (const float*)d_in[9];
    const float* Wref2    = (const float*)d_in[10];
    const float* bref2    = (const float*)d_in[11];
    const float* Vec2     = (const float*)d_in[12];
    const float* dec0     = (const float*)d_in[13];
    float* out = (float*)d_out;

    // workspace layout (bf16 stored as u16)
    char* ws = (char*)d_ws;
    size_t off = 0;
    auto take = [&](size_t bytes) { size_t r = off; off += alignup(bytes); return r; };
    unsigned short* embWb = (unsigned short*)(ws + take((size_t)Esz * Gsz * 2));
    unsigned short* encW  = (unsigned short*)(ws + take((size_t)4 * Hsz * 768 * 2));
    unsigned short* decW  = (unsigned short*)(ws + take((size_t)4 * Hsz * 768 * 2));
    unsigned short* Wq2b  = (unsigned short*)(ws + take((size_t)Hsz * Hsz * 2));
    unsigned short* Wrf2b = (unsigned short*)(ws + take((size_t)Hsz * Hsz * 2));
    unsigned short* xb    = (unsigned short*)(ws + take((size_t)Bsz * Ssz * Gsz * 2));
    unsigned short* emb   = (unsigned short*)(ws + take((size_t)Bsz * Ssz * Esz * 2));
    unsigned short* ref   = (unsigned short*)(ws + take((size_t)Bsz * Ssz * Hsz * 2));
    float*          u2    = (float*)         (ws + take((size_t)Bsz * Ssz * Hsz * 4));
    float*          gates = (float*)         (ws + take((size_t)Bsz * 4 * Hsz * 4));
    float*          cst   = (float*)         (ws + take((size_t)Bsz * Hsz * 4));
    unsigned short* xh    = (unsigned short*)(ws + take((size_t)Bsz * 768 * 2));
    float*          qp    = (float*)         (ws + take((size_t)Bsz * Hsz * 4));
    float*          mask  = (float*)         (ws + take((size_t)Bsz * Ssz * 4));
    (void)ws_size; (void)in_sizes; (void)n_in; (void)out_size;

    const int T = 256;
    auto blks = [](long n, int t) { return (int)((n + t - 1) / t); };

    // ---- init state ----
    zero_f32<<<blks(Bsz * Hsz, T), T, 0, stream>>>(cst, Bsz * Hsz);
    zero_u16<<<blks((long)Bsz * 768, T), T, 0, stream>>>(xh, Bsz * 768);
    zero_f32<<<blks(Bsz * Ssz, T), T, 0, stream>>>(mask, Bsz * Ssz);
    init_out<<<blks(Bsz * Psz + Bsz, T), T, 0, stream>>>(out);

    // ---- weight / input conversion to bf16 ----
    cvt_pack<<<blks((long)Bsz * Ssz * Gsz, T), T, 0, stream>>>(x, xb, Bsz * Ssz, Gsz, Gsz, 0);
    cvt_pack<<<blks((long)Esz * Gsz, T), T, 0, stream>>>(emb_W, embWb, Esz, Gsz, Gsz, 0);
    cvt_pack<<<blks((long)4 * Hsz * Esz, T), T, 0, stream>>>(enc_Wih, encW, 4 * Hsz, Esz, 768, 0);
    cvt_pack<<<blks((long)4 * Hsz * Hsz, T), T, 0, stream>>>(enc_Whh, encW, 4 * Hsz, Hsz, 768, 256);
    cvt_pack<<<blks((long)4 * Hsz * Esz, T), T, 0, stream>>>(dec_Wih, decW, 4 * Hsz, Esz, 768, 0);
    cvt_pack<<<blks((long)4 * Hsz * Hsz, T), T, 0, stream>>>(dec_Whh, decW, 4 * Hsz, Hsz, 768, 256);
    cvt_pack<<<blks((long)Hsz * Hsz, T), T, 0, stream>>>(Wq2, Wq2b, Hsz, Hsz, Hsz, 0);
    cvt_pack<<<blks((long)Hsz * Hsz, T), T, 0, stream>>>(Wref2, Wrf2b, Hsz, Hsz, Hsz, 0);

    // ---- embedding GEMM: emb = x @ emb_W^T  (M=65536, N=256, K=128) ----
    {
        dim3 g(Bsz * Ssz / 64, Esz / 256);
        gemm_bf16<true><<<g, 256, 0, stream>>>(nullptr, emb, xb, Gsz, embWb, nullptr, Esz, Gsz);
    }

    // ---- encoder LSTM (16 steps) ----
    for (int t = 0; t < Ssz; ++t) {
        copy_emb_xh<<<blks((long)Bsz * 32, T), T, 0, stream>>>(emb, xh, t);
        dim3 g(Bsz / 64, 4 * Hsz / 256);
        gemm_bf16<false><<<g, 256, 0, stream>>>(gates, nullptr, xh, 768, encW, enc_b, 4 * Hsz, 768);
        lstm_pw<<<blks((long)Bsz * Hsz, T), T, 0, stream>>>(gates, cst, xh, ref, t);
    }

    // ---- u2 = ref @ Wref2^T + bref2  (M=65536, N=512, K=512) ----
    {
        dim3 g(Bsz * Ssz / 64, Hsz / 256);
        gemm_bf16<false><<<g, 256, 0, stream>>>(u2, nullptr, ref, Hsz, Wrf2b, bref2, Hsz, Hsz);
    }

    // ---- decoder: greedy pointer decode (16 steps) ----
    fill_dec0<<<blks((long)Bsz * Esz, T), T, 0, stream>>>(dec0, xh);
    for (int p = 0; p < Psz; ++p) {
        dim3 g1(Bsz / 64, 4 * Hsz / 256);
        gemm_bf16<false><<<g1, 256, 0, stream>>>(gates, nullptr, xh, 768, decW, dec_b, 4 * Hsz, 768);
        lstm_pw<<<blks((long)Bsz * Hsz, T), T, 0, stream>>>(gates, cst, xh, nullptr, -1);
        dim3 g2(Bsz / 64, Hsz / 256);
        gemm_bf16<false><<<g2, 256, 0, stream>>>(qp, nullptr, xh + 256, 768, Wq2b, bq2, Hsz, Hsz);
        attn_step<<<Bsz / 8, 256, 0, stream>>>(qp, u2, Vec2, mask, emb, xh,
                                               out, out + (size_t)Bsz * Psz, p);
    }
}
